// TimeKernelAttention_61125974557082
// MI455X (gfx1250) — matrix operand order repaired and checked by reading, not compiled, not run on hardware
//
#include <hip/hip_runtime.h>
#include <hip/hip_bf16.h>

typedef __attribute__((ext_vector_type(16))) _Float16 v16h;
typedef __attribute__((ext_vector_type(8)))  _Float16 v8h;
typedef __attribute__((ext_vector_type(8)))  float    v8f;

#define WMMA_F16(A_, B_, C_) \
  __builtin_amdgcn_wmma_f32_16x16x32_f16(false, (A_), false, (B_), (short)0, (C_), false, false)

constexpr int Bc  = 4;
constexpr int Lc  = 1024;
constexpr int DMc = 1024;
constexpr int Hc  = 16;
constexpr int Dc  = 64;
constexpr int TDc = 8;
constexpr int Kc  = 8;

// -------------------- helpers --------------------

__device__ inline v8f v8f_zero() {
  v8f z;
#pragma unroll
  for (int i = 0; i < 8; ++i) z[i] = 0.0f;
  return z;
}

// A-operand gather: elements 0..7 from p0, 8..15 from p1 (two 16B loads)
__device__ inline v16h make_a(const _Float16* p0, const _Float16* p1) {
  union { v16h v; v8h h[2]; } u;
  u.h[0] = *(const v8h*)p0;
  u.h[1] = *(const v8h*)p1;
  return u.v;
}

// reduce across the 16 lanes of each half-wave (xor masks stay within halves)
__device__ inline float hmax16(float v) {
  v = fmaxf(v, __shfl_xor(v, 1, 32));
  v = fmaxf(v, __shfl_xor(v, 2, 32));
  v = fmaxf(v, __shfl_xor(v, 4, 32));
  v = fmaxf(v, __shfl_xor(v, 8, 32));
  return v;
}
__device__ inline float hsum16(float v) {
  v += __shfl_xor(v, 1, 32);
  v += __shfl_xor(v, 2, 32);
  v += __shfl_xor(v, 4, 32);
  v += __shfl_xor(v, 8, 32);
  return v;
}

// -------------------- kernel 1: weight transpose + f16 quantize --------------------
// WqT/WkT/WvT: [h][f][d]  (B-operand: column n = output feature, contiguous in d)
// Wo16: row-major copy (rows of W_O are B columns for out = O * W_O^T)
__global__ void tka_prep_weights(const float* __restrict__ WQ, const float* __restrict__ WK,
                                 const float* __restrict__ WV, const float* __restrict__ WO,
                                 _Float16* __restrict__ WqT, _Float16* __restrict__ WkT,
                                 _Float16* __restrict__ WvT, _Float16* __restrict__ Wo16) {
  int idx = blockIdx.x * 256 + threadIdx.x;
  if (idx < DMc * DMc) Wo16[idx] = (_Float16)WO[idx];
  if (idx < Hc * Dc * Dc) {
    int h = idx >> 12, f = (idx >> 6) & 63, d = idx & 63;
    int src = (h * Dc + d) * Dc + f;   // W[h][d][f] -> WT[h][f][d]
    WqT[idx] = (_Float16)WQ[src];
    WkT[idx] = (_Float16)WK[src];
    WvT[idx] = (_Float16)WV[src];
  }
}

// -------------------- kernel 2: gaussian gating --------------------
__device__ inline float gauss_sum(const float* __restrict__ mu, const float* __restrict__ sig,
                                  const float* __restrict__ w, const float* st, int pb) {
  float g = 0.0f;
#pragma unroll 8
  for (int j = 0; j < TDc * Kc; ++j) {
    float tt = st[j >> 3];          // td = j / K
    float dm = tt - mu[pb + j];
    float s  = sig[pb + j];
    g += w[pb + j] * __expf(-0.5f * dm * dm / (s * s));
  }
  return g;
}

__global__ void tka_gauss(const float* __restrict__ P, const float* __restrict__ T,
                          const float* __restrict__ muQ, const float* __restrict__ sgQ,
                          const float* __restrict__ wQ,  const float* __restrict__ gmQ,
                          const float* __restrict__ muK, const float* __restrict__ sgK,
                          const float* __restrict__ wK,  const float* __restrict__ gmK,
                          const float* __restrict__ muV, const float* __restrict__ sgV,
                          const float* __restrict__ wV,  const float* __restrict__ gmV,
                          _Float16* __restrict__ Qg, _Float16* __restrict__ Kg,
                          _Float16* __restrict__ Vg) {
  const int token = blockIdx.x;                       // b*L + l
  const int hd = blockIdx.y * 256 + threadIdx.x;      // h*D + d
  __shared__ float st[TDc];
  if (threadIdx.x < TDc) st[threadIdx.x] = T[token * TDc + threadIdx.x];
  __syncthreads();
  const int h = hd >> 6, d = hd & 63;
  const int b = token >> 10, l = token & 1023;
  const float p = P[(size_t)token * DMc + hd];
  const int pb = hd * (TDc * Kc);
  const size_t o = (((size_t)(b * Hc + h)) * Lc + l) * Dc + d;   // [b][h][l][d]
  Qg[o] = (_Float16)(gauss_sum(muQ, sgQ, wQ, st, pb) * p + gmQ[hd]);
  Kg[o] = (_Float16)(gauss_sum(muK, sgK, wK, st, pb) * p + gmK[hd]);
  Vg[o] = (_Float16)(gauss_sum(muV, sgV, wV, st, pb) * p + gmV[hd]);
}

// -------------------- kernel 3: per-head projection (WMMA) --------------------
// Qh/Kh: [b][h][l][f] f16 (Q pre-scaled by 1/sqrt(D)); Vt: [b][h][f][l] f16
__global__ void __launch_bounds__(32)
tka_proj(const _Float16* __restrict__ Qg, const _Float16* __restrict__ Kg,
         const _Float16* __restrict__ Vg, const _Float16* __restrict__ WqT,
         const _Float16* __restrict__ WkT, const _Float16* __restrict__ WvT,
         _Float16* __restrict__ Qh, _Float16* __restrict__ Kh, _Float16* __restrict__ Vt) {
  const int lane = threadIdx.x;
  const int lt = blockIdx.x, h = blockIdx.y, b = blockIdx.z;
  const int ln = lane & 15, hi8 = (lane >> 4) * 8, hi16 = (lane >> 4) * 16;
  const size_t headBase = ((size_t)(b * Hc + h) * Lc) * Dc;
  const size_t rowBase  = headBase + (size_t)(lt * 16) * Dc;
  const _Float16* srcs[3] = { Qg + rowBase, Kg + rowBase, Vg + rowBase };
  const _Float16* wts[3]  = { WqT + h * Dc * Dc, WkT + h * Dc * Dc, WvT + h * Dc * Dc };

  for (int q = 0; q < 3; ++q) {
    v8f acc[4];
#pragma unroll
    for (int i = 0; i < 4; ++i) acc[i] = v8f_zero();
    const _Float16* srow = srcs[q] + (size_t)ln * Dc;
#pragma unroll
    for (int kc = 0; kc < 2; ++kc) {
      const int k0 = kc * 32;
      v16h A = make_a(srow + k0 + hi16, srow + k0 + hi16 + 8);
#pragma unroll
      for (int t = 0; t < 4; ++t) {
        const _Float16* bp = wts[q] + (size_t)(t * 16 + ln) * Dc + k0 + hi16;
        acc[t] = WMMA_F16(A, *(const v16h*)bp, acc[t]);
      }
    }
    if (q < 2) {
      _Float16* dst = (q == 0 ? Qh : Kh) + rowBase;
      const float sc = (q == 0) ? 0.125f : 1.0f;   // fold 1/sqrt(D) into Q
#pragma unroll
      for (int t = 0; t < 4; ++t)
#pragma unroll
        for (int r = 0; r < 8; ++r)
          dst[(size_t)(r + hi8) * Dc + t * 16 + ln] = (_Float16)(acc[t][r] * sc);
    } else {
      _Float16* dst = Vt + (size_t)(b * Hc + h) * Dc * Lc;   // [f][l]
      const int lbase = lt * 16;
#pragma unroll
      for (int t = 0; t < 4; ++t)
#pragma unroll
        for (int r = 0; r < 8; ++r)
          dst[(size_t)(t * 16 + ln) * Lc + lbase + r + hi8] = (_Float16)acc[t][r];
    }
  }
}

// -------------------- kernel 4: flash attention (WMMA, 1 wave / 16 rows) --------------------
__global__ void __launch_bounds__(32)
tka_attn(const _Float16* __restrict__ Qh, const _Float16* __restrict__ Kh,
         const _Float16* __restrict__ Vt, _Float16* __restrict__ Obuf) {
  __shared__ _Float16 sP[16 * 32];
  const int lane = threadIdx.x;
  const int lt = blockIdx.x, h = blockIdx.y, b = blockIdx.z;
  const int ln = lane & 15, hi8 = (lane >> 4) * 8, hi16 = (lane >> 4) * 16;
  const size_t headBase = ((size_t)(b * Hc + h) * Lc) * Dc;

  const _Float16* qrow = Qh + headBase + (size_t)(lt * 16 + ln) * Dc;
  v16h qA[2];
  qA[0] = make_a(qrow + hi16, qrow + hi16 + 8);
  qA[1] = make_a(qrow + 32 + hi16, qrow + 32 + hi16 + 8);
  const _Float16* kbase = Kh + headBase;
  const _Float16* vbase = Vt + (size_t)(b * Hc + h) * Dc * Lc;

  v8f o[4];
#pragma unroll
  for (int i = 0; i < 4; ++i) o[i] = v8f_zero();
  float mrow[8], lrow[8];
#pragma unroll
  for (int r = 0; r < 8; ++r) { mrow[r] = -3.0e38f; lrow[r] = 0.0f; }

  for (int mb = 0; mb < Lc / 32; ++mb) {
    const int m0 = mb * 32;
    v8f S0 = v8f_zero(), S1 = v8f_zero();
#pragma unroll
    for (int kc = 0; kc < 2; ++kc) {
      const _Float16* kp0 = kbase + (size_t)(m0 + ln) * Dc + kc * 32 + hi16;
      S0 = WMMA_F16(qA[kc], *(const v16h*)kp0, S0);
      S1 = WMMA_F16(qA[kc], *(const v16h*)(kp0 + 16 * Dc), S1);
    }
    // online softmax update (scores already scaled via Q)
    float alpha[8];
#pragma unroll
    for (int r = 0; r < 8; ++r) {
      float mx = hmax16(fmaxf(S0[r], S1[r]));
      float mn = fmaxf(mrow[r], mx);
      alpha[r] = __expf(mrow[r] - mn);
      mrow[r] = mn;
      float p0 = __expf(S0[r] - mn);
      float p1 = __expf(S1[r] - mn);
      S0[r] = p0; S1[r] = p1;
      lrow[r] = lrow[r] * alpha[r] + hsum16(p0 + p1);
    }
#pragma unroll
    for (int t = 0; t < 4; ++t)
#pragma unroll
      for (int r = 0; r < 8; ++r) o[t][r] *= alpha[r];

    // C-layout f32 -> A-layout f16 via LDS
    __syncthreads();
#pragma unroll
    for (int r = 0; r < 8; ++r) {
      const int row = r + hi8;
      sP[row * 32 + ln]      = (_Float16)S0[r];
      sP[row * 32 + 16 + ln] = (_Float16)S1[r];
    }
    __syncthreads();
    v16h pA = make_a(&sP[ln * 32 + hi16], &sP[ln * 32 + hi16 + 8]);
#pragma unroll
    for (int t = 0; t < 4; ++t) {
      const _Float16* vp = vbase + (size_t)(t * 16 + ln) * Lc + m0 + hi16;
      o[t] = WMMA_F16(pA, *(const v16h*)vp, o[t]);
    }
  }

  // epilogue: normalize, write [b][l][DM] at column h*64
  _Float16* orow = Obuf + ((size_t)(b * Lc + lt * 16)) * DMc + h * Dc;
  float rl[8];
#pragma unroll
  for (int r = 0; r < 8; ++r) rl[r] = 1.0f / lrow[r];
#pragma unroll
  for (int t = 0; t < 4; ++t)
#pragma unroll
    for (int r = 0; r < 8; ++r)
      orow[(size_t)(r + hi8) * DMc + t * 16 + ln] = (_Float16)(o[t][r] * rl[r]);
}

// -------------------- kernel 5: output projection (WMMA) + bias --------------------
__global__ void __launch_bounds__(32)
tka_outproj(const _Float16* __restrict__ Obuf, const _Float16* __restrict__ Wo16,
            const float* __restrict__ bias, float* __restrict__ out) {
  const int lane = threadIdx.x;
  const int mt = blockIdx.x;   // token tile (16 rows)
  const int cg = blockIdx.y;   // 64-column group
  const int ln = lane & 15, hi8 = (lane >> 4) * 8, hi16 = (lane >> 4) * 16;
  const _Float16* arow = Obuf + (size_t)(mt * 16 + ln) * DMc;
  v8f acc[4];
#pragma unroll
  for (int i = 0; i < 4; ++i) acc[i] = v8f_zero();

  for (int k0 = 0; k0 < DMc; k0 += 32) {
    v16h A = make_a(arow + k0 + hi16, arow + k0 + hi16 + 8);
#pragma unroll
    for (int t = 0; t < 4; ++t) {
      const _Float16* bp = Wo16 + (size_t)(cg * 64 + t * 16 + ln) * DMc + k0 + hi16;
      acc[t] = WMMA_F16(A, *(const v16h*)bp, acc[t]);
    }
  }
#pragma unroll
  for (int t = 0; t < 4; ++t) {
    const int col = cg * 64 + t * 16 + ln;
    const float bv = bias[col];
#pragma unroll
    for (int r = 0; r < 8; ++r)
      out[(size_t)(mt * 16 + r + hi8) * DMc + col] = acc[t][r] + bv;
  }
}

// -------------------- launch --------------------
extern "C" void kernel_launch(void* const* d_in, const int* in_sizes, int n_in,
                              void* d_out, int out_size, void* d_ws, size_t ws_size,
                              hipStream_t stream) {
  const float* P   = (const float*)d_in[0];
  const float* T   = (const float*)d_in[1];
  const float* muQ = (const float*)d_in[2];
  const float* sgQ = (const float*)d_in[3];
  const float* wQ  = (const float*)d_in[4];
  const float* gmQ = (const float*)d_in[5];
  const float* WQ  = (const float*)d_in[6];
  const float* muK = (const float*)d_in[7];
  const float* sgK = (const float*)d_in[8];
  const float* wK  = (const float*)d_in[9];
  const float* gmK = (const float*)d_in[10];
  const float* WK  = (const float*)d_in[11];
  const float* muV = (const float*)d_in[12];
  const float* sgV = (const float*)d_in[13];
  const float* wV  = (const float*)d_in[14];
  const float* gmV = (const float*)d_in[15];
  const float* WV  = (const float*)d_in[16];
  const float* WO  = (const float*)d_in[17];
  const float* WOb = (const float*)d_in[18];
  float* out = (float*)d_out;

  // workspace carve-up (halves)
  const size_t n1 = (size_t)Bc * Hc * Lc * Dc;    // 4,194,304
  _Float16* ws = (_Float16*)d_ws;
  _Float16* Qg   = ws;
  _Float16* Kg   = ws + 1 * n1;
  _Float16* Vg   = ws + 2 * n1;
  _Float16* Qh   = ws + 3 * n1;
  _Float16* Kh   = ws + 4 * n1;
  _Float16* Vt   = ws + 5 * n1;
  _Float16* Obuf = ws + 6 * n1;
  _Float16* WqT  = ws + 7 * n1;
  _Float16* WkT  = WqT + (size_t)Hc * Dc * Dc;
  _Float16* WvT  = WkT + (size_t)Hc * Dc * Dc;
  _Float16* Wo16 = WvT + (size_t)Hc * Dc * Dc;

  tka_prep_weights<<<dim3(DMc * DMc / 256), dim3(256), 0, stream>>>(
      WQ, WK, WV, WO, WqT, WkT, WvT, Wo16);

  tka_gauss<<<dim3(Bc * Lc, (Hc * Dc) / 256), dim3(256), 0, stream>>>(
      P, T, muQ, sgQ, wQ, gmQ, muK, sgK, wK, gmK, muV, sgV, wV, gmV, Qg, Kg, Vg);

  tka_proj<<<dim3(Lc / 16, Hc, Bc), dim3(32), 0, stream>>>(
      Qg, Kg, Vg, WqT, WkT, WvT, Qh, Kh, Vt);

  tka_attn<<<dim3(Lc / 16, Hc, Bc), dim3(32), 0, stream>>>(Qh, Kh, Vt, Obuf);

  tka_outproj<<<dim3((Bc * Lc) / 16, DMc / 64), dim3(32), 0, stream>>>(
      Obuf, Wo16, WOb, out);
}
